// TTRFluxLayer_15779709846167
// MI455X (gfx1250) — compile-verified, matrix operand order checked by reading, not run on hardware
//
#include <hip/hip_runtime.h>
#include <hip/hip_bf16.h>
#include <stdint.h>

// ---------------------------------------------------------------------------
// TTR flux layer for MI455X (gfx1250, wave32, WMMA).
//   kernel 1: phi projection  (x @ w1 + b1 -> SiLU -> @ w2 + b2), bf16 WMMA
//   kernel 2: bidirectional causal linear-attention chunked scan, bf16 WMMA,
//             state W^T kept in LDS (f32), K tile staged via Tensor Data Mover.
// Workspace: q_phi / k_phi in bf16, 2 * 24*4096*256*2 B = 96 MB of d_ws.
// ---------------------------------------------------------------------------

typedef __bf16 bf16_t;
typedef __attribute__((ext_vector_type(16))) __bf16 v16bf;
typedef __attribute__((ext_vector_type(8)))  float   v8f;
typedef __attribute__((ext_vector_type(4)))  unsigned int u32x4;
typedef __attribute__((ext_vector_type(8)))  int i32x8;
typedef __attribute__((ext_vector_type(4)))  int i32x4;

static constexpr int kBH   = 24;
static constexpr int kSEQ  = 4096;
static constexpr int kD    = 128;
static constexpr int kF    = 256;
static constexpr int kCHK  = 128;
static constexpr int kNCHK = 32;

// padded LDS row strides (elements); rows stay 16B aligned
static constexpr int LDK = 264;  // 256 + 8
static constexpr int LDD = 136;  // 128 + 8
static constexpr unsigned kSmemBytes = 307200;  // 300 KB of the 320 KB WGP LDS

#if defined(__HIP_DEVICE_COMPILE__) && __has_builtin(__builtin_amdgcn_tensor_load_to_lds) && __has_builtin(__builtin_amdgcn_s_wait_tensorcnt)
#define USE_TDM 1
#else
#define USE_TDM 0
#endif

union FragBF { v16bf v; uint4 q[2]; };

// Load one 16x32 bf16 A/B fragment (per-lane pointer already includes
// row*stride + g*8; two contiguous 16B chunks -> ds_load_b128 / global_load_b128)
__device__ __forceinline__ v16bf frag_ld(const bf16_t* p) {
  FragBF f;
  f.q[0] = *(const uint4*)(p);
  f.q[1] = *(const uint4*)(p + 16);
  return f.v;
}

// Same fragment but sourced from f32 (the LDS-resident W^T state), cvt -> bf16
__device__ __forceinline__ v16bf frag_cvt_f32(const float* p) {
  v16bf a;
#pragma unroll
  for (int j = 0; j < 8; ++j) {
    a[j]     = (bf16_t)p[j];
    a[j + 8] = (bf16_t)p[j + 16];
  }
  return a;
}

__device__ __forceinline__ v8f wmma_bf16(v16bf a, v16bf b, v8f c) {
  return __builtin_amdgcn_wmma_f32_16x16x32_bf16(false, a, false, b, (short)0, c,
                                                 false, false);
}

// fast reciprocal: single v_rcp_f32 (1 ULP; far above the bf16 noise floor)
__device__ __forceinline__ float fast_rcp(float x) {
  return __builtin_amdgcn_rcpf(x);
}

// ---------------------------------------------------------------------------
__global__ void ttr_zero_kernel(float* __restrict__ p, int n) {
  for (int i = blockIdx.x * 256 + threadIdx.x; i < n; i += gridDim.x * 256)
    p[i] = 0.0f;
}

// ---------------------------------------------------------------------------
// phi: grid (48, 4): x = bh*2 + {q,k}, y = quarter of the sequence (8 chunks)
__global__ __launch_bounds__(256) void ttr_phi_kernel(
    const float* __restrict__ qin, const float* __restrict__ kin,
    const float* __restrict__ w1, const float* __restrict__ b1,
    const float* __restrict__ w2, const float* __restrict__ b2,
    bf16_t* __restrict__ qphi, bf16_t* __restrict__ kphi) {
  extern __shared__ char smem[];
  bf16_t* W1T = (bf16_t*)(smem);             // [F][LDD]  w1^T (f x din)   69632 B
  bf16_t* W2T = (bf16_t*)(smem + 69632);     // [F][LDK]  w2^T (fo x f)   135168 B
  bf16_t* XB  = (bf16_t*)(smem + 204800);    // [CHK][LDD] x bf16          34816 B
  bf16_t* HB  = (bf16_t*)(smem + 239616);    // [CHK][LDK] h bf16          67584 B

  const int src = blockIdx.x & 1;
  const int bh  = blockIdx.x >> 1;
  const int tid = threadIdx.x, lane = tid & 31, wv = tid >> 5;
  const int r = lane & 15, g8 = (lane >> 4) << 3;

  const float* X   = (src ? kin : qin) + (size_t)bh * kSEQ * kD;
  bf16_t*      OUT = (src ? kphi : qphi) + (size_t)bh * kSEQ * kF;

  // stage transposed bf16 weights once per block
  for (int it = tid; it < (kD * kF) / 4; it += 256) {
    int din = it >> 6, f0 = (it & 63) << 2;
    float4 raw = *(const float4*)(w1 + din * kF + f0);
    W1T[(f0 + 0) * LDD + din] = (bf16_t)raw.x;
    W1T[(f0 + 1) * LDD + din] = (bf16_t)raw.y;
    W1T[(f0 + 2) * LDD + din] = (bf16_t)raw.z;
    W1T[(f0 + 3) * LDD + din] = (bf16_t)raw.w;
  }
  for (int it = tid; it < (kF * kF) / 4; it += 256) {
    int f = it >> 6, o0 = (it & 63) << 2;
    float4 raw = *(const float4*)(w2 + f * kF + o0);
    W2T[(o0 + 0) * LDK + f] = (bf16_t)raw.x;
    W2T[(o0 + 1) * LDK + f] = (bf16_t)raw.y;
    W2T[(o0 + 2) * LDK + f] = (bf16_t)raw.z;
    W2T[(o0 + 3) * LDK + f] = (bf16_t)raw.w;
  }
  __syncthreads();

  for (int cc = 0; cc < 8; ++cc) {
    const int tb = (blockIdx.y * 8 + cc) * kCHK;

    for (int it = tid; it < (kCHK * kD) / 4; it += 256) {
      int slot = it >> 5, d0 = (it & 31) << 2;
      float4 raw = *(const float4*)(X + (size_t)(tb + slot) * kD + d0);
      bf16_t* xp = XB + slot * LDD + d0;
      xp[0] = (bf16_t)raw.x; xp[1] = (bf16_t)raw.y;
      xp[2] = (bf16_t)raw.z; xp[3] = (bf16_t)raw.w;
    }
    __syncthreads();

    // H = SiLU(X @ w1 + b1): wave wv owns 16-row strip, 16 col tiles, K=128
    {
      const bf16_t* xr = XB + (wv * 16 + r) * LDD + g8;
      v16bf a[4];
#pragma unroll
      for (int kb = 0; kb < 4; ++kb) a[kb] = frag_ld(xr + kb * 32);
#pragma unroll
      for (int nt = 0; nt < 16; ++nt) {
        const bf16_t* wr = W1T + (nt * 16 + r) * LDD + g8;
        v8f acc = {};
#pragma unroll
        for (int kb = 0; kb < 4; ++kb)
          acc = wmma_bf16(a[kb], frag_ld(wr + kb * 32), acc);
        const float bias = b1[nt * 16 + r];
#pragma unroll
        for (int e = 0; e < 8; ++e) {
          float hv = acc[e] + bias;
          // SiLU via v_exp + v_rcp (no IEEE division expansion)
          hv = hv * fast_rcp(1.0f + __expf(-hv));
          HB[(wv * 16 + e + g8) * LDK + nt * 16 + r] = (bf16_t)hv;
        }
      }
    }
    __syncthreads();

    // OUT = H @ w2 + b2 : same strip, K=256
    {
      const bf16_t* hr = HB + (wv * 16 + r) * LDK + g8;
      v16bf a[8];
#pragma unroll
      for (int kb = 0; kb < 8; ++kb) a[kb] = frag_ld(hr + kb * 32);
#pragma unroll
      for (int nt = 0; nt < 16; ++nt) {
        const bf16_t* wr = W2T + (nt * 16 + r) * LDK + g8;
        v8f acc = {};
#pragma unroll
        for (int kb = 0; kb < 8; ++kb)
          acc = wmma_bf16(a[kb], frag_ld(wr + kb * 32), acc);
        const float bias = b2[nt * 16 + r];
#pragma unroll
        for (int e = 0; e < 8; ++e)
          OUT[(size_t)(tb + wv * 16 + e + g8) * kF + nt * 16 + r] =
              (bf16_t)(acc[e] + bias);
      }
    }
    __syncthreads();
  }
}

// ---------------------------------------------------------------------------
// scan: grid (24, 2): one block per (head, direction). 8 waves.
//   per chunk:  S = Q K^T (causal mask)          K-dim 256
//               O = S_bf16 @ V + Q @ W           K-dim 128 / 256
//               W += K^T V                       K-dim 128
__global__ __launch_bounds__(256) void ttr_scan_kernel(
    const float* __restrict__ vin, const bf16_t* __restrict__ qphi,
    const bf16_t* __restrict__ kphi, float* __restrict__ out) {
  extern __shared__ char smem[];
  float*  WT = (float*)(smem);               // [D][LDK] f32  W^T (d x f)  135168 B
  bf16_t* KT = (bf16_t*)(smem + 135168);     // [F][LDD] bf16 K^T (f x t)   69632 B
  bf16_t* VT = (bf16_t*)(smem + 204800);     // [D][LDD] bf16 V^T (d x t)   34816 B
  bf16_t* KC = (bf16_t*)(smem + 239616);     // [CHK][LDK] bf16 K row-major 67584 B
  bf16_t* SB = KC;                           // [CHK][LDD] bf16 S (reuses KC)

  const int bh  = blockIdx.x;
  const int dir = blockIdx.y;
  const int tid = threadIdx.x, lane = tid & 31, wv = tid >> 5;
  const int r = lane & 15, g8 = (lane >> 4) << 3;
  const int m0 = wv * 16;

  const bf16_t* Qg = qphi + (size_t)bh * kSEQ * kF;
  const bf16_t* Kg = kphi + (size_t)bh * kSEQ * kF;
  const float*  Vg = vin + (size_t)bh * kSEQ * kD;
  float*        Og = out + (size_t)bh * kSEQ * kD;

  for (int i = tid; i < kD * LDK; i += 256) WT[i] = 0.0f;
  __syncthreads();

  const uint32_t kc_lds = (uint32_t)(uintptr_t)KC;  // low 32 bits = LDS offset
  (void)kc_lds;

  for (int c = 0; c < kNCHK; ++c) {
    const int s0 = c * kCHK;                              // chunk base slot
    const int t0 = dir ? (kSEQ - (c + 1) * kCHK) : s0;    // ascending token block

    // ---- stage K row-major via TDM (HW row-padding), plus K^T / V^T ----
#if USE_TDM
    if (tid < 32) {
      const uint64_t ga = (uint64_t)(uintptr_t)(Kg + (size_t)t0 * kF);
      u32x4 g0;
      g0[0] = 1u;                                  // count=1, user descriptor
      g0[1] = kc_lds;                              // lds_addr
      g0[2] = (uint32_t)ga;                        // global_addr[31:0]
      g0[3] = (uint32_t)((ga >> 32) & 0x01FFFFFFu) | (2u << 30);  // addr hi, type=2
      i32x8 g1;
      // data_size=2B, pad_enable, pad every 128 dwords (row) by 4 dwords (16B)
      g1[0] = (1 << 16) | (1 << 20) | (6 << 22) | (3 << 25);
      g1[1] = (int)0x80000000u;                    // tensor_dim0 = 32768 elems
      g1[2] = (1 << 16);                           // tensor_dim1 = 1
      g1[3] = (int)0x80000000u;                    // tile_dim0 = 32768
      g1[4] = 1;                                   // tile_dim1 = 1
      g1[5] = 32768;                               // tensor_dim0_stride
      g1[6] = 0; g1[7] = 0;
      i32x4 gz4 = {0, 0, 0, 0};
      i32x8 gz8 = {0, 0, 0, 0, 0, 0, 0, 0};
      __builtin_amdgcn_tensor_load_to_lds(g0, g1, gz4, gz4, gz8, 0);
    }
#else
    for (int it = tid; it < (kCHK * kF) / 8; it += 256) {
      int row = it >> 5, f0 = (it & 31) << 3;
      *(uint4*)(KC + row * LDK + f0) =
          *(const uint4*)(Kg + (size_t)(t0 + row) * kF + f0);
    }
#endif
    for (int it = tid; it < (kCHK * kF) / 8; it += 256) {   // K^T in slot order
      int slot = it >> 5, f0 = (it & 31) << 3;
      int tok = dir ? (kSEQ - 1 - (s0 + slot)) : (s0 + slot);
      union { uint4 u; bf16_t h[8]; } cv;
      cv.u = *(const uint4*)(Kg + (size_t)tok * kF + f0);
#pragma unroll
      for (int j = 0; j < 8; ++j) KT[(f0 + j) * LDD + slot] = cv.h[j];
    }
    for (int it = tid; it < (kCHK * kD) / 4; it += 256) {   // V^T in slot order
      int slot = it >> 5, d0 = (it & 31) << 2;
      int tok = dir ? (kSEQ - 1 - (s0 + slot)) : (s0 + slot);
      float4 raw = *(const float4*)(Vg + (size_t)tok * kD + d0);
      VT[(d0 + 0) * LDD + slot] = (bf16_t)raw.x;
      VT[(d0 + 1) * LDD + slot] = (bf16_t)raw.y;
      VT[(d0 + 2) * LDD + slot] = (bf16_t)raw.z;
      VT[(d0 + 3) * LDD + slot] = (bf16_t)raw.w;
    }
#if USE_TDM
    if (tid < 32) __builtin_amdgcn_s_wait_tensorcnt(0);
#endif
    __syncthreads();

    // ---- S = Q K^T over K=256, wave wv owns query rows [m0, m0+16) ----
    const int qtok = dir ? (kSEQ - 1 - (s0 + m0 + r)) : (s0 + m0 + r);
    const bf16_t* qrow = Qg + (size_t)qtok * kF + g8;
    v16bf qfr[8];
#pragma unroll
    for (int kb = 0; kb < 8; ++kb) qfr[kb] = frag_ld(qrow + kb * 32);

    v8f Sacc[8];
#pragma unroll
    for (int nt = 0; nt < 8; ++nt) {
      const int ks = nt * 16 + r;                      // key slot this lane holds
      const int krow = dir ? (kCHK - 1 - ks) : ks;     // LDS row of that slot
      const bf16_t* kp = KC + krow * LDK + g8;
      v8f acc = {};
#pragma unroll
      for (int kb = 0; kb < 8; ++kb)
        acc = wmma_bf16(qfr[kb], frag_ld(kp + kb * 32), acc);
      Sacc[nt] = acc;
    }
    __syncthreads();  // all waves done reading KC -> overwrite with S_bf16

    // causal mask (inclusive, j <= i) + cvt to bf16, slot-ordered
#pragma unroll
    for (int nt = 0; nt < 8; ++nt) {
      const int j = nt * 16 + r;
#pragma unroll
      for (int e = 0; e < 8; ++e) {
        const int i = m0 + e + g8;
        SB[i * LDD + j] = (bf16_t)((j <= i) ? Sacc[nt][e] : 0.0f);
      }
    }
    __syncthreads();

    // ---- O = S @ V  (only lower-triangular k-steps) + Q @ W ----
    v8f Oacc[8];
    {
      v8f z = {};
#pragma unroll
      for (int nt = 0; nt < 8; ++nt) Oacc[nt] = z;
    }
    const int nkb = (wv >> 1) + 1;
    const bf16_t* srow = SB + (m0 + r) * LDD + g8;
    for (int kb = 0; kb < nkb; ++kb) {
      v16bf a = frag_ld(srow + kb * 32);
#pragma unroll
      for (int nt = 0; nt < 8; ++nt)
        Oacc[nt] = wmma_bf16(
            a, frag_ld(VT + (nt * 16 + r) * LDD + g8 + kb * 32), Oacc[nt]);
    }
#pragma unroll
    for (int kb = 0; kb < 8; ++kb) {
#pragma unroll
      for (int nt = 0; nt < 8; ++nt)
        Oacc[nt] = wmma_bf16(
            qfr[kb], frag_cvt_f32(WT + (nt * 16 + r) * LDK + g8 + kb * 32),
            Oacc[nt]);
    }
    __syncthreads();  // everyone done reading W before it is updated

    // ---- W += K^T V : wave wv owns f-tiles {2wv, 2wv+1} x all 8 d-tiles ----
#pragma unroll
    for (int h = 0; h < 2; ++h) {
      const int fm0 = (wv * 2 + h) * 16;
      const bf16_t* ktrow = KT + (fm0 + r) * LDD + g8;
      v16bf afr[4];
#pragma unroll
      for (int kb = 0; kb < 4; ++kb) afr[kb] = frag_ld(ktrow + kb * 32);
#pragma unroll
      for (int nt = 0; nt < 8; ++nt) {
        float* wp = WT + (nt * 16 + r) * LDK + fm0 + g8;
        v8f acc;
#pragma unroll
        for (int e = 0; e < 8; ++e) acc[e] = wp[e];
#pragma unroll
        for (int kb = 0; kb < 4; ++kb)
          acc = wmma_bf16(
              afr[kb], frag_ld(VT + (nt * 16 + r) * LDD + g8 + kb * 32), acc);
#pragma unroll
        for (int e = 0; e < 8; ++e) wp[e] = acc[e];
      }
    }

    // ---- scale by 1/(pos+1) and accumulate (fwd + flipped rev) ----
#pragma unroll
    for (int nt = 0; nt < 8; ++nt) {
#pragma unroll
      for (int e = 0; e < 8; ++e) {
        const int i = m0 + e + g8;
        const int sg = s0 + i;                         // position in scan order
        const int tok = dir ? (kSEQ - 1 - sg) : sg;    // original token
        const float val = Oacc[nt][e] * fast_rcp((float)(sg + 1));
        __hip_atomic_fetch_add(&Og[(size_t)tok * kD + nt * 16 + r], val,
                               __ATOMIC_RELAXED, __HIP_MEMORY_SCOPE_AGENT);
      }
    }
    __syncthreads();  // LDS reuse fence before next chunk staging
  }
}

// ---------------------------------------------------------------------------
extern "C" void kernel_launch(void* const* d_in, const int* in_sizes, int n_in,
                              void* d_out, int out_size, void* d_ws,
                              size_t ws_size, hipStream_t stream) {
  (void)in_sizes; (void)n_in; (void)ws_size;
  const float* q  = (const float*)d_in[0];
  const float* k  = (const float*)d_in[1];
  const float* v  = (const float*)d_in[2];
  const float* w1 = (const float*)d_in[3];
  const float* b1 = (const float*)d_in[4];
  const float* w2 = (const float*)d_in[5];
  const float* b2 = (const float*)d_in[6];
  float* out = (float*)d_out;

  bf16_t* qphi = (bf16_t*)d_ws;                          // 48 MB
  bf16_t* kphi = qphi + (size_t)kBH * kSEQ * kF;         // 48 MB

  ttr_zero_kernel<<<1024, 256, 0, stream>>>(out, out_size);
  ttr_phi_kernel<<<dim3(2 * kBH, 4), 256, kSmemBytes, stream>>>(
      q, k, w1, b1, w2, b2, qphi, kphi);
  ttr_scan_kernel<<<dim3(kBH, 2), 256, kSmemBytes, stream>>>(v, qphi, kphi, out);
}
